// E3NN_Conv_0to2_52510270161534
// MI455X (gfx1250) — matrix operand order repaired and checked by reading, not compiled, not run on hardware
//
#include <hip/hip_runtime.h>
#include <math.h>

typedef __attribute__((ext_vector_type(16))) _Float16 v16h;
typedef __attribute__((ext_vector_type(4)))  _Float16 v4h;
typedef __attribute__((ext_vector_type(8)))  float    v8f;

#define NN   20000
#define EE   320000
#define INF  16
#define OUTF 16
#define HID  32
#define EPSV 1e-6f
#define INV_SD 0.25f   // 1/sqrt(E/N) = 1/sqrt(16)

#define SH_C0  0.28209479177387814f
#define SH_C1  0.4886025119029199f
#define SH_C2A 1.0925484305920792f
#define SH_C2B 0.31539156525252005f
#define SH_C2C 0.5462742152960396f

#define NTILES (EE/16)            // 20000 edge tiles of 16
#define NSTEPS 17                 // K = 17*32 = 544 (512 outer-product + bias + pad)
#define NLAYER 3
#define BSTEPS (NSTEPS*NLAYER)    // 51
#define BTAB_HALVES (BSTEPS*32*16)// 26112 f16 = 52224 bytes

#define CONV_BLOCKS 402           // 402*8 = 3216 waves, divisible by 3
#define CONV_WAVES  (CONV_BLOCKS*8)
#define CONV_W3     (CONV_WAVES/3)

__device__ __forceinline__ float shflf(float v, int sl) {
  return __int_as_float(__builtin_amdgcn_ds_bpermute(sl << 2, __float_as_int(v)));
}
__device__ __forceinline__ int shfli(int v, int sl) {
  return __builtin_amdgcn_ds_bpermute(sl << 2, v);
}

// Build one A fragment: a[j] = h0*hs[j], a[8+j] = h1*hs[j]
// with h0,h1 = relu(d*rw1[2t{,+1}] + rb1[2t{,+1}]) in f16.
__device__ __forceinline__ v16h buildA(v4h q, _Float16 dh, const _Float16 hsh[8]) {
  const _Float16 zh = (_Float16)0.f;
  _Float16 h0 = q[0] * dh + q[2];
  _Float16 h1 = q[1] * dh + q[3];
  h0 = (h0 > zh) ? h0 : zh;
  h1 = (h1 > zh) ? h1 : zh;
  v16h a;
#pragma unroll
  for (int j = 0; j < 8; ++j) { a[j] = h0 * hsh[j]; a[8 + j] = h1 * hsh[j]; }
  return a;
}

// ---------------- Kernel 1: h = x @ w_in  (N,16)@(16,16) ----------------
__global__ void k_h(const float* __restrict__ x, const float* __restrict__ w_in,
                    float* __restrict__ h) {
  int i = blockIdx.x * blockDim.x + threadIdx.x;
  if (i >= NN * OUTF) return;
  int n = i >> 4, c = i & 15;
  float s = 0.f;
#pragma unroll
  for (int f = 0; f < INF; ++f) s = fmaf(x[n * INF + f], w_in[f * OUTF + c], s);
  h[i] = s;
}

// ---- Kernel 2: build f16 B-table in the ISA 16-bit B-fragment layout ----
// W2'[k*16+i, o] = rw2_l[k, 16o+i]; row 512+i = rb2_l[16o+i]; rows 528.. = 0
// Step t covers K-slice [32t,32t+32); lane ln -> N=ln&15, h=ln>>4; element j:
// k=2t+(j>>3), i=8h+(j&7).
__global__ void k_btab(const float* __restrict__ rw2_0, const float* __restrict__ rb2_0,
                       const float* __restrict__ rw2_1, const float* __restrict__ rb2_1,
                       const float* __restrict__ rw2_2, const float* __restrict__ rb2_2,
                       _Float16* __restrict__ btab) {
  int i = blockIdx.x * blockDim.x + threadIdx.x;
  if (i >= BTAB_HALVES) return;
  int j  = i & 15;
  int ln = (i >> 4) & 31;
  int st = i >> 9;                 // 0..50
  int l  = st / NSTEPS;
  int t  = st % NSTEPS;
  int kk = 2 * t + (j >> 3);
  int ii = 8 * (ln >> 4) + (j & 7);
  int n  = ln & 15;
  const float* rw2 = (l == 0) ? rw2_0 : ((l == 1) ? rw2_1 : rw2_2);
  const float* rb2 = (l == 0) ? rb2_0 : ((l == 1) ? rb2_1 : rb2_2);
  float v = 0.f;
  if (kk < HID)       v = rw2[kk * 256 + n * 16 + ii];
  else if (kk == HID) v = rb2[n * 16 + ii];
  btab[i] = (_Float16)v;
}

// ---- Kernel 2b: packed radial-weight table, f16 ----
// Q[l*17+t] = (rw1[2t], rw1[2t+1], rb1[2t], rb1[2t+1]); t==16 -> (0,0,1,0)
// so hr'[32] = relu(0*d+1) = 1 (bias row), hr'[33] = relu(0) = 0 (pad row).
__global__ void k_qtab(const float* __restrict__ rw1_0, const float* __restrict__ rb1_0,
                       const float* __restrict__ rw1_1, const float* __restrict__ rb1_1,
                       const float* __restrict__ rw1_2, const float* __restrict__ rb1_2,
                       _Float16* __restrict__ qtab) {
  int i = threadIdx.x;
  if (i >= BSTEPS) return;
  int l = i / NSTEPS, t = i % NSTEPS;
  const float* rw1 = (l == 0) ? rw1_0 : ((l == 1) ? rw1_1 : rw1_2);
  const float* rb1 = (l == 0) ? rb1_0 : ((l == 1) ? rb1_1 : rb1_2);
  float a = 0.f, b = 0.f, c = 1.f, d = 0.f;
  if (t < 16) { a = rw1[2*t]; b = rw1[2*t+1]; c = rb1[2*t]; d = rb1[2*t+1]; }
  qtab[i*4+0] = (_Float16)a; qtab[i*4+1] = (_Float16)b;
  qtab[i*4+2] = (_Float16)c; qtab[i*4+3] = (_Float16)d;
}

// ---------------- Kernel 3: edge-tile WMMA conv + scatter ----------------
// One layer per wave; that layer's 17 B-fragments stay register-resident.
// A fragments ping-pong between two buffers; an empty asm that reads BOTH
// the just-consumed buffer and the freshly built one (a) forces the fake
// use to schedule after the build and (b) keeps the old buffer live across
// it, so RA keeps the buffers in distinct registers -> no WMMA->VALU WAR
// hazard nop padding.
__global__ void __launch_bounds__(256) k_conv(
    const int* __restrict__ eidx, const float* __restrict__ eattr,
    const float* __restrict__ h,
    const _Float16* __restrict__ gBtab, const _Float16* __restrict__ qtab,
    float* __restrict__ acc0, float* __restrict__ acc1, float* __restrict__ acc2) {
  int tid   = threadIdx.x;
  int lane  = tid & 31;
  int half  = lane >> 4;       // 0: i in [0,8), 1: i in [8,16)
  int el    = lane & 15;       // edge-in-tile this lane owns metadata for
  int half8 = half * 8;
  int o     = el;              // C-matrix N index for this lane

  int gw = blockIdx.x * (blockDim.x >> 5) + (tid >> 5);
  int l  = gw % NLAYER;        // fixed layer for this wave
  int t0 = gw / NLAYER;

  // Load this layer's B fragments into registers (once per wave).
  v16h Bf[NSTEPS];
  {
    const _Float16* bp = gBtab + (l * NSTEPS * 32 + lane) * 16;
#pragma unroll
    for (int t = 0; t < NSTEPS; ++t) Bf[t] = *(const v16h*)(bp + t * 512);
  }
  // Packed radial weights for this layer (f16, 34 VGPRs).
  v4h Q[NSTEPS];
  {
    const v4h* qp = (const v4h*)(qtab + l * NSTEPS * 4);
#pragma unroll
    for (int t = 0; t < NSTEPS; ++t) Q[t] = qp[t];
  }

  for (int tile = t0; tile < NTILES; tile += CONV_W3) {
    int e   = tile * 16 + el;
    int src = eidx[e];
    int dst = eidx[EE + e];
    float ax = eattr[e * 3 + 0], ay = eattr[e * 3 + 1], az = eattr[e * 3 + 2];
    float dlen = sqrtf(ax * ax + ay * ay + az * az);
    float inv  = 1.0f / (dlen + EPSV);
    float ux = ax * inv, uy = ay * inv, uz = az * inv;
    _Float16 dh = (_Float16)dlen;

    // hs slice for this lane: h[src, half*8 .. half*8+7] in f16
    _Float16 hsh[8];
    {
      const float4* hp = (const float4*)(h + src * 16 + half8);
      float4 h0 = hp[0], h1 = hp[1];
      hsh[0] = (_Float16)h0.x; hsh[1] = (_Float16)h0.y;
      hsh[2] = (_Float16)h0.z; hsh[3] = (_Float16)h0.w;
      hsh[4] = (_Float16)h1.x; hsh[5] = (_Float16)h1.y;
      hsh[6] = (_Float16)h1.z; hsh[7] = (_Float16)h1.w;
    }

    // Ping-pong pipelined WMMA chain.
    v16h a0 = buildA(Q[0], dh, hsh);
    v16h a1;
    v8f c = {0.f, 0.f, 0.f, 0.f, 0.f, 0.f, 0.f, 0.f};
#pragma unroll
    for (int t = 0; t < NSTEPS; ++t) {
      if ((t & 1) == 0) {
        c = __builtin_amdgcn_wmma_f32_16x16x32_f16(false, a0, false, Bf[t],
                                                   (short)0, c, false, false);
        if (t + 1 < NSTEPS) {
          a1 = buildA(Q[t + 1], dh, hsh);
          asm volatile("" :: "v"(a0), "v"(a1));  // a0 live across a1's build
        } else {
          asm volatile("" :: "v"(a0));
        }
      } else {
        c = __builtin_amdgcn_wmma_f32_16x16x32_f16(false, a1, false, Bf[t],
                                                   (short)0, c, false, false);
        if (t + 1 < NSTEPS) {
          a0 = buildA(Q[t + 1], dh, hsh);
          asm volatile("" :: "v"(a1), "v"(a0));  // a1 live across a0's build
        } else {
          asm volatile("" :: "v"(a1));
        }
      }
      // Pin ALU/WMMA ordering per step; let VMEM/DS ops move for latency hiding.
      __builtin_amdgcn_sched_barrier(0x3F0);
    }

    // scatter: row r of C is edge (r + half*8); fetch its dst/u cross-lane
    if (l == 0) {
#pragma unroll
      for (int r = 0; r < 8; ++r) {
        int sl = r + half8;
        int dd = shfli(dst, sl);
        atomicAdd(acc0 + dd * 16 + o, c[r] * (SH_C0 * INV_SD));
      }
    } else if (l == 1) {
#pragma unroll
      for (int r = 0; r < 8; ++r) {
        int sl = r + half8;
        int dd = shfli(dst, sl);
        float sx = shflf(ux, sl), sy = shflf(uy, sl), sz = shflf(uz, sl);
        float m = c[r] * (SH_C1 * INV_SD);
        float* p = acc1 + (dd * 16 + o) * 3;
        atomicAdd(p + 0, m * sx); atomicAdd(p + 1, m * sy); atomicAdd(p + 2, m * sz);
      }
    } else {
#pragma unroll
      for (int r = 0; r < 8; ++r) {
        int sl = r + half8;
        int dd = shfli(dst, sl);
        float sx = shflf(ux, sl), sy = shflf(uy, sl), sz = shflf(uz, sl);
        float m = c[r] * INV_SD;
        float y0 = SH_C2A * sx * sy, y1 = SH_C2A * sy * sz;
        float y2 = SH_C2B * (3.f * sz * sz - 1.f);
        float y3 = SH_C2A * sx * sz, y4 = SH_C2C * (sx * sx - sy * sy);
        float* p = acc2 + (dd * 16 + o) * 5;
        atomicAdd(p + 0, m * y0); atomicAdd(p + 1, m * y1); atomicAdd(p + 2, m * y2);
        atomicAdd(p + 3, m * y3); atomicAdd(p + 4, m * y4);
      }
    }
  }
}

// ------------- Kernel 4: per-channel sum of squares (for RMS) -------------
__global__ void k_reduce(const float* __restrict__ acc0, const float* __restrict__ acc1,
                         const float* __restrict__ acc2, float* __restrict__ ss) {
  __shared__ float sbuf[48];
  int tid = threadIdx.x;
  if (tid < 48) sbuf[tid] = 0.f;
  __syncthreads();
  int i = blockIdx.x * blockDim.x + tid;
  if (i < NN * 16) {
    int cch = i & 15;
    float a = acc0[i];
    float s0 = a * a, s1 = 0.f, s2 = 0.f;
#pragma unroll
    for (int m = 0; m < 3; ++m) { float v = acc1[i * 3 + m]; s1 += v * v; }
#pragma unroll
    for (int m = 0; m < 5; ++m) { float v = acc2[i * 5 + m]; s2 += v * v; }
    atomicAdd(&sbuf[cch], s0);
    atomicAdd(&sbuf[16 + cch], s1);
    atomicAdd(&sbuf[32 + cch], s2);
  }
  __syncthreads();
  if (tid < 48) atomicAdd(&ss[tid], sbuf[tid]);
}

// --------- Kernel 5: field-norm, w_out mix, relu / sigmoid gates ---------
__global__ void k_final(const float* __restrict__ acc0, const float* __restrict__ acc1,
                        const float* __restrict__ acc2, const float* __restrict__ ss,
                        const float* __restrict__ w_out0, const float* __restrict__ w_out1,
                        const float* __restrict__ w_out2,
                        const float* __restrict__ b0, const float* __restrict__ b1,
                        const float* __restrict__ b2, float* __restrict__ out) {
  int n = blockIdx.x * blockDim.x + threadIdx.x;
  if (n >= NN) return;
  float tn0[16], tn1[48], tn2[80];
#pragma unroll
  for (int c = 0; c < 16; ++c) {
    float i0 = 1.f / (sqrtf(ss[c] / (float)NN) + EPSV);
    float i1 = 1.f / (sqrtf(ss[16 + c] / (float)NN) + EPSV);
    float i2 = 1.f / (sqrtf(ss[32 + c] / (float)NN) + EPSV);
    tn0[c] = acc0[n * 16 + c] * i0;
#pragma unroll
    for (int m = 0; m < 3; ++m) tn1[c * 3 + m] = acc1[(n * 16 + c) * 3 + m] * i1;
#pragma unroll
    for (int m = 0; m < 5; ++m) tn2[c * 5 + m] = acc2[(n * 16 + c) * 5 + m] * i2;
  }
  float* out0 = out;
  float* out1 = out + NN * 16;
  float* out2 = out + NN * 64;
  for (int d = 0; d < 16; ++d) {
    float o0 = 0.f, o1[3] = {0, 0, 0}, o2[5] = {0, 0, 0, 0, 0};
#pragma unroll
    for (int c = 0; c < 16; ++c) {
      float w0 = w_out0[c * 16 + d], w1 = w_out1[c * 16 + d], w2 = w_out2[c * 16 + d];
      o0 = fmaf(tn0[c], w0, o0);
#pragma unroll
      for (int m = 0; m < 3; ++m) o1[m] = fmaf(tn1[c * 3 + m], w1, o1[m]);
#pragma unroll
      for (int m = 0; m < 5; ++m) o2[m] = fmaf(tn2[c * 5 + m], w2, o2[m]);
    }
    out0[n * 16 + d] = fmaxf(o0 + b0[d], 0.f);
    float n1 = sqrtf(o1[0] * o1[0] + o1[1] * o1[1] + o1[2] * o1[2]);
    float g1 = 1.f / (1.f + expf(-(n1 + b1[d])));
#pragma unroll
    for (int m = 0; m < 3; ++m) out1[(n * 16 + d) * 3 + m] = o1[m] * g1;
    float n2 = 0.f;
#pragma unroll
    for (int m = 0; m < 5; ++m) n2 += o2[m] * o2[m];
    n2 = sqrtf(n2);
    float g2 = 1.f / (1.f + expf(-(n2 + b2[d])));
#pragma unroll
    for (int m = 0; m < 5; ++m) out2[(n * 16 + d) * 5 + m] = o2[m] * g2;
  }
}

extern "C" void kernel_launch(void* const* d_in, const int* in_sizes, int n_in,
                              void* d_out, int out_size, void* d_ws, size_t ws_size,
                              hipStream_t stream) {
  const float* x     = (const float*)d_in[0];
  const int*   eidx  = (const int*)d_in[1];
  const float* eattr = (const float*)d_in[2];
  const float* w_in  = (const float*)d_in[3];
  const float* rw1[3] = {(const float*)d_in[4],  (const float*)d_in[8],  (const float*)d_in[12]};
  const float* rb1[3] = {(const float*)d_in[5],  (const float*)d_in[9],  (const float*)d_in[13]};
  const float* rw2[3] = {(const float*)d_in[6],  (const float*)d_in[10], (const float*)d_in[14]};
  const float* rb2[3] = {(const float*)d_in[7],  (const float*)d_in[11], (const float*)d_in[15]};
  const float* w_out0 = (const float*)d_in[16];
  const float* b_nl0  = (const float*)d_in[17];
  const float* w_out1 = (const float*)d_in[18];
  const float* b_nl1  = (const float*)d_in[19];
  const float* w_out2 = (const float*)d_in[20];
  const float* b_nl2  = (const float*)d_in[21];
  float* out = (float*)d_out;

  char* w = (char*)d_ws;
  float*    d_h    = (float*)(w + 0);                 // 1,280,000 B
  _Float16* d_btab = (_Float16*)(w + 1280000);        //    52,224 B
  _Float16* d_qtab = (_Float16*)(w + 1332224);        //       408 B (padded to 512)
  float*    d_acc0 = (float*)(w + 1332736);           // 1,280,000 B
  float*    d_acc1 = (float*)(w + 2612736);           // 3,840,000 B
  float*    d_acc2 = (float*)(w + 6452736);           // 6,400,000 B
  float*    d_ss   = (float*)(w + 12852736);          //       192 B

  // zero accumulators + ss (contiguous region)
  hipMemsetAsync(w + 1332736, 0, 11520192, stream);

  k_h<<<(NN * OUTF + 255) / 256, 256, 0, stream>>>(x, w_in, d_h);
  k_btab<<<(BTAB_HALVES + 255) / 256, 256, 0, stream>>>(
      rw2[0], rb2[0], rw2[1], rb2[1], rw2[2], rb2[2], d_btab);
  k_qtab<<<1, 64, 0, stream>>>(rw1[0], rb1[0], rw1[1], rb1[1], rw1[2], rb1[2], d_qtab);
  k_conv<<<CONV_BLOCKS, 256, 0, stream>>>(eidx, eattr, d_h, d_btab, d_qtab,
                                          d_acc0, d_acc1, d_acc2);
  k_reduce<<<(NN * 16 + 255) / 256, 256, 0, stream>>>(d_acc0, d_acc1, d_acc2, d_ss);
  k_final<<<(NN + 127) / 128, 128, 0, stream>>>(d_acc0, d_acc1, d_acc2, d_ss,
                                                w_out0, w_out1, w_out2,
                                                b_nl0, b_nl1, b_nl2, out);
}
